// SACActor_67388036874465
// MI455X (gfx1250) — compile-verified, matrix operand order checked by reading, not run on hardware
//
#include <hip/hip_runtime.h>

// ---------------------------------------------------------------------------
// Types for CDNA5 WMMA (wave32, 16x16x32 bf16 -> f32)
// ---------------------------------------------------------------------------
typedef __bf16 bf16;
typedef __attribute__((ext_vector_type(16))) __bf16 v16bf;
typedef __attribute__((ext_vector_type(8)))  __bf16 v8bf;
typedef __attribute__((ext_vector_type(8)))  float  v8f;

union U16 { v16bf v; v8bf h[2]; };

__device__ inline v8f splat8(float x) { v8f r = {x,x,x,x,x,x,x,x}; return r; }

// Model dims
#define B_   256
#define S_   256
#define IND  64
#define H_   512
#define G4   2048
#define TOK  (B_*S_)

// ---------------------------------------------------------------------------
// Pack a [Ksrc, N] f32 row-major weight into per-lane WMMA B fragments (bf16).
// Fragment addressing: frag(kt, nt) = packed[ ((kt+ktOff)*N/16 + nt)*32*16 .. ]
// lane L holds column n = nt*16 + (L&15); element e maps to
//   k = kt*32 + (L>>4)*8 + e + (e>=8 ? 8 : 0)
// ---------------------------------------------------------------------------
__global__ void k_pack_b(const float* __restrict__ w, bf16* __restrict__ packed,
                         int N, int ktOff) {
  int lane = threadIdx.x;                 // 32 threads
  int NT = N >> 4;
  int nt = blockIdx.x % NT;
  int kt = blockIdx.x / NT;
  int g8 = (lane >> 4) * 8;
  int n  = nt * 16 + (lane & 15);
  bf16* out = packed + (((size_t)(kt + ktOff) * NT + nt) * 32 + lane) * 16;
#pragma unroll
  for (int e = 0; e < 16; ++e) {
    int k = kt * 32 + g8 + e + ((e >= 8) ? 8 : 0);
    out[e] = (bf16)w[(size_t)k * N + n];
  }
}

// ---------------------------------------------------------------------------
// Generic WMMA GEMM: C[M,N] = A_f32[M,K] @ Bpacked_bf16[K,N] + bias.
// Block = 256 threads (8 waves); tile 128(M) x 64(N); wave w owns M rows
// w*16..w*16+15 and all 4 N-tiles. A converted f32->bf16 through LDS.
// ---------------------------------------------------------------------------
template <int OUT_BF16>
__global__ void __launch_bounds__(256)
k_gemm(const float* __restrict__ A, const bf16* __restrict__ Bp,
       const float* __restrict__ bias, void* __restrict__ Cout,
       int M, int N, int K) {
  __shared__ bf16 As[128 * 32];
  int tid  = threadIdx.x;
  int wv   = tid >> 5, lane = tid & 31;
  int m0   = blockIdx.x * 128;
  int n0   = blockIdx.y * 64;
  int NT   = N >> 4;

  v8f acc[4];
#pragma unroll
  for (int nt = 0; nt < 4; ++nt)
    acc[nt] = splat8(bias[n0 + nt * 16 + (lane & 15)]);

  int mrow = wv * 16 + (lane & 15);
  int g8   = (lane >> 4) * 8;

  for (int k0 = 0; k0 < K; k0 += 32) {
    __syncthreads();
#pragma unroll
    for (int i = 0; i < 16; ++i) {            // stage 128x32 A tile
      int e = tid + i * 256;
      int r = e >> 5, c = e & 31;
      As[r * 32 + c] = (bf16)A[(size_t)(m0 + r) * K + (k0 + c)];
    }
    __syncthreads();

    U16 af;
    af.h[0] = *(const v8bf*)&As[mrow * 32 + g8];
    af.h[1] = *(const v8bf*)&As[mrow * 32 + 16 + g8];
    int kt = k0 >> 5;
#pragma unroll
    for (int nt = 0; nt < 4; ++nt) {
      int ntG = (n0 >> 4) + nt;
      v16bf bf = *(const v16bf*)(Bp + (((size_t)kt * NT + ntG) * 32 + lane) * 16);
      acc[nt] = __builtin_amdgcn_wmma_f32_16x16x32_bf16(
          false, af.v, false, bf, (short)0, acc[nt], false, false);
    }
  }

#pragma unroll
  for (int nt = 0; nt < 4; ++nt)
#pragma unroll
    for (int v = 0; v < 8; ++v) {
      int m = m0 + wv * 16 + (lane >> 4) * 8 + v;
      int n = n0 + nt * 16 + (lane & 15);
      if (m < M) {
        if (OUT_BF16) ((bf16*)Cout)[(size_t)m * N + n] = (bf16)acc[nt][v];
        else          ((float*)Cout)[(size_t)m * N + n] = acc[nt][v];
      }
    }
}

// ---------------------------------------------------------------------------
// LayerNorm stats per row (mean / rstd), 256 threads per row.
// ---------------------------------------------------------------------------
__global__ void k_ln_stats(const float* __restrict__ x, float* __restrict__ mean,
                           float* __restrict__ rstd, int dim) {
  int row = blockIdx.x;
  const float* xr = x + (size_t)row * dim;
  float s = 0.f, s2 = 0.f;
  for (int j = threadIdx.x; j < dim; j += blockDim.x) {
    float v = xr[j]; s += v; s2 += v * v;
  }
  __shared__ float sh[8], sh2[8];
  for (int o = 16; o; o >>= 1) { s += __shfl_down(s, o, 32); s2 += __shfl_down(s2, o, 32); }
  if ((threadIdx.x & 31) == 0) { sh[threadIdx.x >> 5] = s; sh2[threadIdx.x >> 5] = s2; }
  __syncthreads();
  if (threadIdx.x == 0) {
    float t = 0.f, t2 = 0.f;
    int nw = blockDim.x >> 5;
    for (int i = 0; i < nw; ++i) { t += sh[i]; t2 += sh2[i]; }
    float m = t / dim;
    float var = t2 / dim - m * m;
    mean[row] = m;
    rstd[row] = rsqrtf(var + 1e-5f);
  }
}

__global__ void k_ln_apply(float* __restrict__ x, const float* __restrict__ mean,
                           const float* __restrict__ rstd, const float* __restrict__ g,
                           const float* __restrict__ b, int dim) {
  size_t idx = (size_t)blockIdx.x * blockDim.x + threadIdx.x;
  size_t row = idx / dim;
  int j = (int)(idx % dim);
  x[idx] = g[j] * (x[idx] - mean[row]) * rstd[row] + b[j];
}

// ---------------------------------------------------------------------------
// sLSTM scan. One block per 16 batch rows; 512 threads = 16 waves (wave32).
// Per timestep:
//   A = [ LN(cur[:,t,:]) | h ]  (16 x 1024 bf16, in LDS)
//   g = A @ Wpacked([wx;wh], 1024x2048) + bias   (WMMA; 8 N-tiles per wave,
//       64 acc VGPRs/lane -> stays inside the direct 256-VGPR window)
//   gates -> c,n,h update; residual into cur; h -> A for t+1.
// Dynamic LDS: A 32KB + g 128KB + c 32KB + n 32KB = 224KB (<320KB/WGP).
// ---------------------------------------------------------------------------
__global__ void __launch_bounds__(512, 1)
k_slstm(float* __restrict__ cur,
        const float* __restrict__ mean, const float* __restrict__ rstd,
        const float* __restrict__ lng, const float* __restrict__ lnb,
        const bf16* __restrict__ Wp,      // packed [32 ktiles][128 ntiles][32][16]
        const float* __restrict__ gbias,  // [2048]
        float* __restrict__ h_out, float* __restrict__ c_out,
        float* __restrict__ n_out) {
  extern __shared__ char smem[];
  bf16*  As = (bf16*)smem;                          // 16 x 1024
  float* gS = (float*)(smem + 16 * 1024 * 2);       // 16 x 2048
  float* cS = (float*)(smem + 16 * 1024 * 2 + 16 * 2048 * 4);  // 16 x 512
  float* nS = cS + 16 * 512;                        // 16 x 512

  int tid = threadIdx.x, wv = tid >> 5, lane = tid & 31;
  int b0 = blockIdx.x * 16;

  for (int i = tid; i < 16 * 512; i += 512) {
    cS[i] = 0.f; nS[i] = 1.f;
    int r = i >> 9, j = i & 511;
    As[r * 1024 + 512 + j] = (bf16)0.f;
  }
  __syncthreads();

  int mrow   = lane & 15;
  int g8     = (lane >> 4) * 8;
  int ntBase = wv * 8;                 // 16 waves x 8 N-tiles = 128
  float bv[8];
#pragma unroll
  for (int nt = 0; nt < 8; ++nt)
    bv[nt] = gbias[(ntBase + nt) * 16 + (lane & 15)];

  for (int t = 0; t < S_; ++t) {
    // ---- phase A: xn = LN(cur[:,t,:]) -> A[:, 0:512]
    for (int i = tid; i < 16 * 512; i += 512) {
      int r = i >> 9, j = i & 511;
      size_t tok = (size_t)(b0 + r) * S_ + t;
      float xv = cur[tok * H_ + j];
      As[r * 1024 + j] = (bf16)(lng[j] * (xv - mean[tok]) * rstd[tok] + lnb[j]);
    }
    __syncthreads();

    // ---- phase B: g = A @ W + bias (each wave: 8 N-tiles, K=1024)
    v8f acc[8];
#pragma unroll
    for (int nt = 0; nt < 8; ++nt) acc[nt] = splat8(bv[nt]);

    for (int kt = 0; kt < 32; ++kt) {
      U16 af;
      const bf16* ar = &As[mrow * 1024 + kt * 32 + g8];
      af.h[0] = *(const v8bf*)ar;
      af.h[1] = *(const v8bf*)(ar + 16);
      const bf16* bp = Wp + (((size_t)kt * 128 + ntBase) * 32 + lane) * 16;
      if (kt < 31)
        __builtin_prefetch(Wp + (((size_t)(kt + 1) * 128 + ntBase) * 32 + lane) * 16, 0, 0);
#pragma unroll
      for (int nt = 0; nt < 8; ++nt) {
        v16bf bf = *(const v16bf*)(bp + (size_t)nt * 32 * 16);
        acc[nt] = __builtin_amdgcn_wmma_f32_16x16x32_bf16(
            false, af.v, false, bf, (short)0, acc[nt], false, false);
      }
    }
#pragma unroll
    for (int nt = 0; nt < 8; ++nt)
#pragma unroll
      for (int v = 0; v < 8; ++v) {
        int m = (lane >> 4) * 8 + v;
        int n = (ntBase + nt) * 16 + (lane & 15);
        gS[m * 2048 + n] = acc[nt][v];
      }
    __syncthreads();

    // ---- phase C: gates + state update + residual
    for (int i = tid; i < 16 * 512; i += 512) {
      int r = i >> 9, j = i & 511;
      float gi = gS[r * 2048 + j];
      float gf = gS[r * 2048 + 512 + j];
      float gz = gS[r * 2048 + 1024 + j];
      float go = gS[r * 2048 + 1536 + j];
      float iv = expf(gi), fv = expf(gf);
      float zv = tanhf(gz);
      float ov = 1.f / (1.f + expf(-go));
      float c  = fv * cS[i] + iv * zv;
      float nn = fv * nS[i] + iv;
      float h  = ov * (c / nn);
      cS[i] = c; nS[i] = nn;
      As[r * 1024 + 512 + j] = (bf16)h;
      size_t tok = (size_t)(b0 + r) * S_ + t;
      cur[tok * H_ + j] += h;
      if (t == S_ - 1) {
        size_t o = (size_t)(b0 + r) * H_ + j;
        h_out[o] = h; c_out[o] = c; n_out[o] = nn;
      }
    }
    __syncthreads();
  }
}

// ---------------------------------------------------------------------------
// q at last timestep: q[b,n] = bias[n] + cur[b, S-1, :] . Wq[:, n]
// ---------------------------------------------------------------------------
__global__ void k_qlast(const float* __restrict__ cur, const float* __restrict__ w,
                        const float* __restrict__ bias, float* __restrict__ out) {
  int b = blockIdx.x, n = threadIdx.x;   // 512 threads
  const float* xr = cur + ((size_t)b * S_ + (S_ - 1)) * H_;
  float a = bias[n];
  for (int k = 0; k < H_; ++k) a += xr[k] * w[(size_t)k * H_ + n];
  out[(size_t)b * H_ + n] = a;
}

// ---------------------------------------------------------------------------
// Pooled attention: per (b, head) softmax(q_last . K / 8) @ V
// ---------------------------------------------------------------------------
__global__ void k_attn(const float* __restrict__ q, const bf16* __restrict__ kb,
                       const bf16* __restrict__ vb, float* __restrict__ ctx) {
  __shared__ float sc[256];
  __shared__ float qv[64];
  __shared__ float red[8];
  int b = blockIdx.x >> 3, hh = blockIdx.x & 7;
  int tid = threadIdx.x;
  if (tid < 64) qv[tid] = q[(size_t)b * H_ + hh * 64 + tid];
  __syncthreads();

  const bf16* kr = kb + ((size_t)(b * S_ + tid) * H_ + hh * 64);
  float d = 0.f;
#pragma unroll
  for (int e = 0; e < 64; ++e) d += qv[e] * (float)kr[e];
  d *= 0.125f;  // 1/sqrt(64)

  float mx = d;
  for (int o = 16; o; o >>= 1) mx = fmaxf(mx, __shfl_xor(mx, o, 32));
  if ((tid & 31) == 0) red[tid >> 5] = mx;
  __syncthreads();
  mx = red[0];
#pragma unroll
  for (int i = 1; i < 8; ++i) mx = fmaxf(mx, red[i]);
  float e = expf(d - mx);
  __syncthreads();
  float sm = e;
  for (int o = 16; o; o >>= 1) sm += __shfl_xor(sm, o, 32);
  if ((tid & 31) == 0) red[tid >> 5] = sm;
  __syncthreads();
  sm = 0.f;
#pragma unroll
  for (int i = 0; i < 8; ++i) sm += red[i];
  sc[tid] = e / sm;
  __syncthreads();

  if (tid < 64) {
    float a = 0.f;
    for (int s = 0; s < S_; ++s)
      a += sc[s] * (float)vb[(size_t)(b * S_ + s) * H_ + hh * 64 + tid];
    ctx[(size_t)b * H_ + hh * 64 + tid] = a;
  }
}

__global__ void k_oproj(const float* __restrict__ x, const float* __restrict__ w,
                        const float* __restrict__ bias, float* __restrict__ out) {
  int b = blockIdx.x, n = threadIdx.x;   // 512 threads
  const float* xr = x + (size_t)b * H_;
  float a = bias[n];
  for (int k = 0; k < H_; ++k) a += xr[k] * w[(size_t)k * H_ + n];
  out[(size_t)b * H_ + n] = a;
}

// ---------------------------------------------------------------------------
// Row MLP: out = gelu(LN(x1@W[0:K1] + x2@W[K1:K1+K2] + bias)); N = blockDim.x
// ---------------------------------------------------------------------------
__global__ void k_row2lin_ln_gelu(const float* __restrict__ x1, int K1,
                                  const float* __restrict__ x2, int K2, int N,
                                  const float* __restrict__ w, const float* __restrict__ bias,
                                  const float* __restrict__ g, const float* __restrict__ bvec,
                                  float* __restrict__ out) {
  __shared__ float red[8];
  int b = blockIdx.x, j = threadIdx.x;
  float a = bias[j];
  const float* r1 = x1 + (size_t)b * K1;
  for (int k = 0; k < K1; ++k) a += r1[k] * w[(size_t)k * N + j];
  if (x2) {
    const float* r2 = x2 + (size_t)b * K2;
    for (int k = 0; k < K2; ++k) a += r2[k] * w[(size_t)(K1 + k) * N + j];
  }
  int nw = N >> 5;
  float s = a;
  for (int o = 16; o; o >>= 1) s += __shfl_xor(s, o, 32);
  if ((j & 31) == 0) red[j >> 5] = s;
  __syncthreads();
  float m = 0.f;
  for (int i = 0; i < nw; ++i) m += red[i];
  m /= N;
  __syncthreads();
  float d = a - m, s2 = d * d;
  for (int o = 16; o; o >>= 1) s2 += __shfl_xor(s2, o, 32);
  if ((j & 31) == 0) red[j >> 5] = s2;
  __syncthreads();
  float var = 0.f;
  for (int i = 0; i < nw; ++i) var += red[i];
  var /= N;
  float xn = g[j] * d * rsqrtf(var + 1e-5f) + bvec[j];
  out[(size_t)b * N + j] = xn * 0.5f * (1.f + erff(xn * 0.70710678118f));
}

__global__ void k_heads(const float* __restrict__ f2, const float* __restrict__ muw,
                        const float* __restrict__ mub, const float* __restrict__ lsw,
                        const float* __restrict__ lsb, float* __restrict__ out) {
  int b = blockIdx.x, j = threadIdx.x;
  const float* xr = f2 + (size_t)b * 128;
  if (j < 3) {
    float a = mub[j];
    for (int k = 0; k < 128; ++k) a += xr[k] * muw[k * 3 + j];
    out[b * 3 + j] = a;
  } else if (j < 6) {
    int jj = j - 3;
    float a = lsb[jj];
    for (int k = 0; k < 128; ++k) a += xr[k] * lsw[k * 3 + jj];
    out[768 + b * 3 + jj] = fminf(2.0f, fmaxf(-20.0f, a));
  }
}

// ---------------------------------------------------------------------------
// Host launcher. Input order: JAX pytree flatten (dict keys sorted).
// ---------------------------------------------------------------------------
extern "C" void kernel_launch(void* const* d_in, const int* in_sizes, int n_in,
                              void* d_out, int out_size, void* d_ws, size_t ws_size,
                              hipStream_t stream) {
  (void)in_sizes; (void)n_in; (void)out_size; (void)ws_size;
  const float* info  = (const float*)d_in[0];
  const float* akb   = (const float*)d_in[1];
  const float* akw   = (const float*)d_in[2];
  const float* aob   = (const float*)d_in[3];
  const float* aow   = (const float*)d_in[4];
  const float* aqb   = (const float*)d_in[5];
  const float* aqw   = (const float*)d_in[6];
  const float* avb   = (const float*)d_in[7];
  const float* avw   = (const float*)d_in[8];
  const float* bb1b  = (const float*)d_in[9];
  const float* bb1w  = (const float*)d_in[10];
  const float* bb2b  = (const float*)d_in[11];
  const float* bb2w  = (const float*)d_in[12];
  const float* bln1b = (const float*)d_in[13];
  const float* bln1g = (const float*)d_in[14];
  const float* bln2b = (const float*)d_in[15];
  const float* bln2g = (const float*)d_in[16];
  const float* i1b   = (const float*)d_in[17];
  const float* i1w   = (const float*)d_in[18];
  const float* i2b   = (const float*)d_in[19];
  const float* i2w   = (const float*)d_in[20];
  const float* iln1b = (const float*)d_in[21];
  const float* iln1g = (const float*)d_in[22];
  const float* iln2b = (const float*)d_in[23];
  const float* iln2g = (const float*)d_in[24];
  const float* innb  = (const float*)d_in[25];
  const float* inng  = (const float*)d_in[26];
  const float* ipb   = (const float*)d_in[27];
  const float* ipw   = (const float*)d_in[28];
  const float* l_b[2]   = {(const float*)d_in[29], (const float*)d_in[34]};
  const float* l_lnb[2] = {(const float*)d_in[30], (const float*)d_in[35]};
  const float* l_lng[2] = {(const float*)d_in[31], (const float*)d_in[36]};
  const float* l_wh[2]  = {(const float*)d_in[32], (const float*)d_in[37]};
  const float* l_wx[2]  = {(const float*)d_in[33], (const float*)d_in[38]};
  const float* lsb   = (const float*)d_in[39];
  const float* lsw   = (const float*)d_in[40];
  const float* mub   = (const float*)d_in[41];
  const float* muw   = (const float*)d_in[42];
  const float* x     = (const float*)d_in[43];
  float* out = (float*)d_out;

  // Workspace carve
  char* p = (char*)d_ws;
  auto carve = [&](size_t bytes) {
    char* r = p;
    p += (bytes + 255) & ~(size_t)255;
    return r;
  };
  float* cur   = (float*)carve((size_t)TOK * H_ * 4);    // 128 MB
  bf16*  kbuf  = (bf16*)carve((size_t)TOK * H_ * 2);     // 64 MB
  bf16*  vbuf  = (bf16*)carve((size_t)TOK * H_ * 2);     // 64 MB
  float* meanb = (float*)carve((size_t)TOK * 4);
  float* rstdb = (float*)carve((size_t)TOK * 4);
  bf16*  wxh0  = (bf16*)carve((size_t)1024 * G4 * 2);    // 4 MB
  bf16*  wxh1  = (bf16*)carve((size_t)1024 * G4 * 2);
  bf16*  wip   = (bf16*)carve((size_t)IND * H_ * 2);
  bf16*  wkp   = (bf16*)carve((size_t)H_ * H_ * 2);
  bf16*  wvp   = (bf16*)carve((size_t)H_ * H_ * 2);
  float* qlast = (float*)carve((size_t)B_ * H_ * 4);
  float* ctx   = (float*)carve((size_t)B_ * H_ * 4);
  float* ctx2  = (float*)carve((size_t)B_ * H_ * 4);
  float* ieA   = (float*)carve((size_t)B_ * 64 * 4);
  float* ieB   = (float*)carve((size_t)B_ * 64 * 4);
  float* f1    = (float*)carve((size_t)B_ * 256 * 4);
  float* f2    = (float*)carve((size_t)B_ * 128 * 4);
  bf16* wxh[2] = {wxh0, wxh1};

  // 1) Pack weights into WMMA B-fragment layout (bf16)
  k_pack_b<<<dim3((IND / 32) * (H_ / 16)), 32, 0, stream>>>(ipw, wip, H_, 0);
  for (int l = 0; l < 2; ++l) {
    k_pack_b<<<dim3(16 * 128), 32, 0, stream>>>(l_wx[l], wxh[l], G4, 0);   // K rows 0..511
    k_pack_b<<<dim3(16 * 128), 32, 0, stream>>>(l_wh[l], wxh[l], G4, 16);  // K rows 512..1023
  }
  k_pack_b<<<dim3(16 * 32), 32, 0, stream>>>(akw, wkp, H_, 0);
  k_pack_b<<<dim3(16 * 32), 32, 0, stream>>>(avw, wvp, H_, 0);

  // 2) Input projection + input LayerNorm
  k_gemm<0><<<dim3(TOK / 128, H_ / 64), 256, 0, stream>>>(x, wip, ipb, cur, TOK, H_, IND);
  k_ln_stats<<<TOK, 256, 0, stream>>>(cur, meanb, rstdb, H_);
  k_ln_apply<<<(TOK * H_) / 256, 256, 0, stream>>>(cur, meanb, rstdb, inng, innb, H_);

  // 3) Two sLSTM layers (batch-parallel WMMA scan, 16 waves/block)
  size_t smem = (size_t)16 * 1024 * 2 + (size_t)16 * 2048 * 4 + (size_t)16 * 512 * 4 * 2;
  for (int l = 0; l < 2; ++l) {
    k_ln_stats<<<TOK, 256, 0, stream>>>(cur, meanb, rstdb, H_);
    k_slstm<<<B_ / 16, 512, smem, stream>>>(
        cur, meanb, rstdb, l_lng[l], l_lnb[l], wxh[l], l_b[l],
        out + 1536 + (size_t)l * B_ * H_,            // hs
        out + 1536 + 262144 + (size_t)l * B_ * H_,   // cs
        out + 1536 + 524288 + (size_t)l * B_ * H_);  // ns
  }

  // 4) Attention pooled at last timestep
  k_gemm<1><<<dim3(TOK / 128, H_ / 64), 256, 0, stream>>>(cur, wkp, akb, kbuf, TOK, H_, H_);
  k_gemm<1><<<dim3(TOK / 128, H_ / 64), 256, 0, stream>>>(cur, wvp, avb, vbuf, TOK, H_, H_);
  k_qlast<<<B_, H_, 0, stream>>>(cur, aqw, aqb, qlast);
  k_attn<<<B_ * 8, 256, 0, stream>>>(qlast, kbuf, vbuf, ctx);
  k_oproj<<<B_, H_, 0, stream>>>(ctx, aow, aob, ctx2);

  // 5) Info encoder + trunk + heads
  k_row2lin_ln_gelu<<<B_, 64, 0, stream>>>(info, 13, nullptr, 0, 64, i1w, i1b, iln1g, iln1b, ieA);
  k_row2lin_ln_gelu<<<B_, 64, 0, stream>>>(ieA, 64, nullptr, 0, 64, i2w, i2b, iln2g, iln2b, ieB);
  k_row2lin_ln_gelu<<<B_, 256, 0, stream>>>(ctx2, 512, ieB, 64, 256, bb1w, bb1b, bln1g, bln1b, f1);
  k_row2lin_ln_gelu<<<B_, 128, 0, stream>>>(f1, 256, nullptr, 0, 128, bb2w, bb2b, bln2g, bln2b, f2);
  k_heads<<<B_, 64, 0, stream>>>(f2, muw, mub, lsw, lsb, out);
}
